// CustomOpenLmAttn_27247272526207
// MI455X (gfx1250) — matrix operand order repaired and checked by reading, not compiled, not run on hardware
//
#include <hip/hip_runtime.h>
#include <math.h>

// ---------------- problem constants ----------------
#define BATCH 2
#define SEQ   8192
#define DMODEL 2048
#define NHEAD 16
#define HDIM  128
#define NPROJ 7
#define BLKSZ 256
#define NSAMP 256
#define NBLK  (SEQ / BLKSZ)          // 32
#define BH    (BATCH * NHEAD)        // 32
#define MROWS (BATCH * SEQ)          // 16384
#define LOGW  3.4657359028f          // log(8192/256)
#define SCALE 0.08838834764831845f   // 1/sqrt(128)

typedef __bf16 v16bf __attribute__((ext_vector_type(16)));
typedef float  v8f   __attribute__((ext_vector_type(8)));

union FragBF { v16bf v; uint4 q[2]; };

// ---- optional CDNA5 async global->LDS path (guarded: always compiles) ----
#if defined(__has_builtin)
#if __has_builtin(__builtin_amdgcn_global_load_async_to_lds_b128)
#define HAVE_ASYNC_LDS 1
#endif
#if __has_builtin(__builtin_amdgcn_s_wait_asynccnt)
#define HAVE_WAIT_ASYNC 1
#endif
#endif

#ifdef HAVE_ASYNC_LDS
typedef int v4i_t __attribute__((ext_vector_type(4)));
typedef __attribute__((address_space(1))) v4i_t* gv4i_p;   // global (AS1)
typedef __attribute__((address_space(3))) v4i_t* lv4i_p;   // LDS (AS3)
__device__ __forceinline__ void async_g2l_b128(const void* g, void* l) {
    // build AS1 ptr from full 64-bit generic address; AS3 ptr from the low
    // 32 bits of the generic shared address (the LDS byte offset)
    gv4i_p gp = (gv4i_p)(unsigned long long)(g);
    lv4i_p lp = (lv4i_p)(unsigned int)(unsigned long long)(l);
    __builtin_amdgcn_global_load_async_to_lds_b128(gp, lp, 0, 0);
}
#endif

__device__ __forceinline__ void wait_async_lds() {
#ifdef HAVE_WAIT_ASYNC
    __builtin_amdgcn_s_wait_asynccnt(0);
#endif
}

__device__ __forceinline__ v8f v8f_zero() {
    v8f z;
#pragma unroll
    for (int i = 0; i < 8; ++i) z[i] = 0.0f;
    return z;
}

__device__ __forceinline__ v8f wmma_bf16(v16bf a, v16bf b, v8f c) {
    // D = A(16x32 bf16) x B(32x16 bf16) + C(16x16 f32)
    return __builtin_amdgcn_wmma_f32_16x16x32_bf16(false, a, false, b,
                                                   (short)0, c, false, false);
}

// ---------------- 1) f32 -> bf16 cast ----------------
__global__ void cast_f32_bf16_kernel(const float* __restrict__ in,
                                     __bf16* __restrict__ out, int n) {
    int i = blockIdx.x * blockDim.x + threadIdx.x;
    int stride = gridDim.x * blockDim.x;
    for (; i < n; i += stride) out[i] = (__bf16)in[i];
}

// ---------------- 2/7) bf16 WMMA GEMM: C(MxN) = A(MxK) * B(KxN) ----------------
// 128x128 block tile, 256 threads (8 waves as 2x4), wave tile 64x32.
template <typename OUT_T>
__global__ __launch_bounds__(256) void gemm_bf16_kernel(
    const __bf16* __restrict__ A, const __bf16* __restrict__ B,
    OUT_T* __restrict__ C, int M, int N, int K) {
    __shared__ __bf16 As[128][40];   // row-major [m][k], padded
    __shared__ __bf16 Bt[128][40];   // transposed [n][k], padded

    const int tid = threadIdx.x;
    const int w = tid >> 5, l = tid & 31;
    const int m = l & 15, hi = l >> 4;
    const int bm = blockIdx.y * 128, bn = blockIdx.x * 128;
    const int wm = (w >> 2) * 64, wn = (w & 3) * 32;

    v8f acc[4][2];
#pragma unroll
    for (int mt = 0; mt < 4; ++mt)
#pragma unroll
        for (int nt = 0; nt < 2; ++nt) acc[mt][nt] = v8f_zero();

    for (int k0 = 0; k0 < K; k0 += 32) {
        __syncthreads();
        {   // A tile: 128 rows x 32 halves; thread -> 16 halves (32B)
            int row = tid >> 1, part = tid & 1;
            const __bf16* src = A + (size_t)(bm + row) * K + k0 + part * 16;
#ifdef HAVE_ASYNC_LDS
            async_g2l_b128(src,     &As[row][part * 16]);
            async_g2l_b128(src + 8, &As[row][part * 16 + 8]);
#else
            *(uint4*)&As[row][part * 16]     = ((const uint4*)src)[0];
            *(uint4*)&As[row][part * 16 + 8] = ((const uint4*)src)[1];
#endif
        }
        {   // B tile transposed: rows k0..k0+31, cols bn..bn+127
            int kk = tid & 31, np = tid >> 5;   // np in 0..7 -> 16 cols
            const __bf16* src = B + (size_t)(k0 + kk) * N + bn + np * 16;
            __bf16 tmp[16];
            *(uint4*)&tmp[0] = ((const uint4*)src)[0];
            *(uint4*)&tmp[8] = ((const uint4*)src)[1];
#pragma unroll
            for (int i = 0; i < 16; ++i) Bt[np * 16 + i][kk] = tmp[i];
        }
        // warm GL2 with the next K-tile while WMMAs run on this one
        if (k0 + 32 < K) {
            __builtin_prefetch(A + (size_t)(bm + (tid >> 1)) * K + k0 + 32, 0, 1);
            __builtin_prefetch(B + (size_t)(k0 + 32 + (tid & 31)) * N + bn + (tid >> 5) * 16, 0, 1);
        }
        wait_async_lds();
        __syncthreads();

        FragBF bf0, bf1;
        bf0.q[0] = *(const uint4*)&Bt[wn + m][16 * hi];
        bf0.q[1] = *(const uint4*)&Bt[wn + m][16 * hi + 8];
        bf1.q[0] = *(const uint4*)&Bt[wn + 16 + m][16 * hi];
        bf1.q[1] = *(const uint4*)&Bt[wn + 16 + m][16 * hi + 8];
#pragma unroll
        for (int mt = 0; mt < 4; ++mt) {
            FragBF af;
            af.q[0] = *(const uint4*)&As[wm + mt * 16 + m][16 * hi];
            af.q[1] = *(const uint4*)&As[wm + mt * 16 + m][16 * hi + 8];
            acc[mt][0] = wmma_bf16(af.v, bf0.v, acc[mt][0]);
            acc[mt][1] = wmma_bf16(af.v, bf1.v, acc[mt][1]);
        }
    }
    // C layout: VGPR r -> row r+8*hi, col l&15
#pragma unroll
    for (int mt = 0; mt < 4; ++mt)
#pragma unroll
        for (int nt = 0; nt < 2; ++nt)
#pragma unroll
            for (int r = 0; r < 8; ++r) {
                int gm = bm + wm + mt * 16 + r + 8 * hi;
                int gn = bn + wn + nt * 16 + m;
                C[(size_t)gm * N + gn] = (OUT_T)acc[mt][nt][r];
            }
}

// ---------------- 3) RoPE + reshape (B,S,3,H,HD) -> (B,H,S,HD) ----------------
__global__ __launch_bounds__(256) void rope_kernel(
    const __bf16* __restrict__ qkv, __bf16* __restrict__ Qo,
    __bf16* __restrict__ Ko, __bf16* __restrict__ Vo) {
    int t = blockIdx.x * 256 + threadIdx.x;     // B*S*H*64 = 16,777,216 threads
    int d = t & 63;
    int h = (t >> 6) & 15;
    int s = (t >> 10) & 8191;
    int b = t >> 23;
    size_t row = (size_t)(b * SEQ + s) * (3 * DMODEL);
    size_t o = ((size_t)(b * NHEAD + h) * SEQ + s) * HDIM;

    float inv_freq = __powf(10000.0f, -(float)d * (1.0f / 64.0f));
    float fr = (float)s * inv_freq;
    float sn, cs;
    __sincosf(fr, &sn, &cs);

    float q1 = (float)qkv[row + h * HDIM + d];
    float q2 = (float)qkv[row + h * HDIM + d + 64];
    Qo[o + d]      = (__bf16)(q1 * cs - q2 * sn);
    Qo[o + d + 64] = (__bf16)(q2 * cs + q1 * sn);

    float k1 = (float)qkv[row + DMODEL + h * HDIM + d];
    float k2 = (float)qkv[row + DMODEL + h * HDIM + d + 64];
    Ko[o + d]      = (__bf16)(k1 * cs - k2 * sn);
    Ko[o + d + 64] = (__bf16)(k2 * cs + k1 * sn);

    Vo[o + d]      = qkv[row + 2 * DMODEL + h * HDIM + d];
    Vo[o + d + 64] = qkv[row + 2 * DMODEL + h * HDIM + d + 64];
}

// ---------------- 4) LSH hash: groups 0..31 = q, 32..63 = k ----------------
__global__ __launch_bounds__(256) void lsh_hash_kernel(
    const __bf16* __restrict__ Qb, const __bf16* __restrict__ Kb,
    const float* __restrict__ proj, int* __restrict__ hashes) {
    int t = blockIdx.x * 256 + threadIdx.x;     // 64 * 8192 threads
    if (t >= 64 * SEQ) return;
    int s = t & (SEQ - 1);
    int g = t >> 13;
    const __bf16* src = (g < BH) ? Qb + ((size_t)g * SEQ + s) * HDIM
                                 : Kb + ((size_t)(g - BH) * SEQ + s) * HDIM;
    float acc[NPROJ];
#pragma unroll
    for (int p = 0; p < NPROJ; ++p) acc[p] = 0.0f;
    for (int d = 0; d < HDIM; ++d) {
        float x = (float)src[d];
#pragma unroll
        for (int p = 0; p < NPROJ; ++p) acc[p] += x * proj[d * NPROJ + p];
    }
    int hsh = 0;
#pragma unroll
    for (int p = 0; p < NPROJ; ++p)
        if (acc[p] > 0.0f) hsh |= (1 << p);
    hashes[t] = hsh;
}

// ---------------- 5) stable argsort via bitonic on (hash<<13 | idx) ----------------
__global__ __launch_bounds__(1024) void bitonic_sort_kernel(
    const int* __restrict__ hashes, int* __restrict__ sidx) {
    __shared__ unsigned int key[SEQ];   // 32 KB
    const int g = blockIdx.x;           // 0..63
    const int* hp = hashes + (size_t)g * SEQ;
    for (int i = threadIdx.x; i < SEQ; i += 1024)
        key[i] = ((unsigned)hp[i] << 13) | (unsigned)i;
    __syncthreads();
    for (int kk = 2; kk <= SEQ; kk <<= 1) {
        for (int j = kk >> 1; j > 0; j >>= 1) {
            for (int i = threadIdx.x; i < SEQ; i += 1024) {
                int ixj = i ^ j;
                if (ixj > i) {
                    bool up = (i & kk) == 0;
                    unsigned a = key[i], b = key[ixj];
                    if ((a > b) == up) { key[i] = b; key[ixj] = a; }
                }
            }
            __syncthreads();
        }
    }
    int* op = sidx + (size_t)g * SEQ;
    for (int i = threadIdx.x; i < SEQ; i += 1024)
        op[i] = (int)(key[i] & 0x1FFFu);
}

// ---------------- 6) fused hyper-attention (flash-style online softmax) --------
// grid: BH * NBLK blocks; 512 threads = 16 waves; wave owns 16 q rows.
// 16 kv chunks of 32: chunks 0..7 = this sorted block (bias 0),
// chunks 8..15 = sampled set (bias LOGW, -inf on block collision).
__global__ __launch_bounds__(512) void hyper_attn_kernel(
    const __bf16* __restrict__ Qb, const __bf16* __restrict__ Kb,
    const __bf16* __restrict__ Vb, const int* __restrict__ sidx_q,
    const int* __restrict__ sidx_k, const int* __restrict__ sampled,
    __bf16* __restrict__ Out) {
    __shared__ __bf16 Kc[32][136];        // kv-chunk rows x HDIM, padded
    __shared__ __bf16 Vt[128][40];        // V transposed: [d][kv], padded
    __shared__ __bf16 Pst[16][16][40];    // per-wave P staging, padded
    __shared__ float  biasCol[32];

    const int blk = blockIdx.x;
    const int qb = blk & (NBLK - 1);
    const int bh = blk >> 5;
    const int b = bh >> 4, h = bh & 15;

    const int tid = threadIdx.x;
    const int w = tid >> 5, l = tid & 31;
    const int m = l & 15, hi = l >> 4;

    const __bf16* Qh = Qb + (size_t)bh * SEQ * HDIM;
    const __bf16* Kh = Kb + (size_t)bh * SEQ * HDIM;
    const __bf16* Vh = Vb + (size_t)bh * SEQ * HDIM;
    const int* sq_base = sidx_q + (size_t)bh * SEQ + qb * BLKSZ;
    const int* sk_base = sidx_k + (size_t)bh * SEQ;

    // resident Q A-fragments (16 rows x 128 d, 4 K-steps)
    FragBF qf[4];
    {
        int srow = sq_base[w * 16 + m];
        const __bf16* qrow = Qh + (size_t)srow * HDIM;
#pragma unroll
        for (int ks = 0; ks < 4; ++ks) {
            qf[ks].q[0] = *(const uint4*)(qrow + ks * 32 + 16 * hi);
            qf[ks].q[1] = *(const uint4*)(qrow + ks * 32 + 16 * hi + 8);
        }
    }
    int sqr[8];
#pragma unroll
    for (int r = 0; r < 8; ++r) sqr[r] = sq_base[w * 16 + r + 8 * hi];

    v8f oacc[8];
#pragma unroll
    for (int ot = 0; ot < 8; ++ot) oacc[ot] = v8f_zero();
    float mstate[8], lstate[8];
#pragma unroll
    for (int r = 0; r < 8; ++r) { mstate[r] = -3.0e38f; lstate[r] = 0.0f; }

    for (int c = 0; c < 16; ++c) {
        __syncthreads();
        {   // cooperative gather of K chunk (row-major) and V chunk (transposed)
            int row = tid >> 4, part = tid & 15;    // 32 rows x 16 parts (8 halves)
            int spos;
            float bias;
            if (c < 8) {
                spos = qb * BLKSZ + c * 32 + row;
                bias = 0.0f;
            } else {
                int j = (c - 8) * 32 + row;
                int sv = sampled[(bh << 8) + j];     // position in sorted domain
                spos = sv;
                bias = ((sv >> 8) == qb) ? -1.0e30f : LOGW;
            }
            int grow = sk_base[spos];
            const __bf16* kr = Kh + (size_t)grow * HDIM + part * 8;
#ifdef HAVE_ASYNC_LDS
            async_g2l_b128(kr, &Kc[row][part * 8]);
#else
            *(uint4*)&Kc[row][part * 8] = *(const uint4*)kr;
#endif
            const __bf16* vr = Vh + (size_t)grow * HDIM + part * 8;
            __bf16 vt[8];
            *(uint4*)vt = *(const uint4*)vr;
#pragma unroll
            for (int i = 0; i < 8; ++i) Vt[part * 8 + i][row] = vt[i];
            if (part == 0) biasCol[row] = bias;
        }
        wait_async_lds();
        __syncthreads();

        // scores S = Q * K^T (16 x 32), two 16-col tiles
        v8f s0 = v8f_zero(), s1 = v8f_zero();
#pragma unroll
        for (int ks = 0; ks < 4; ++ks) {
            FragBF k0, k1;
            k0.q[0] = *(const uint4*)&Kc[m][ks * 32 + 16 * hi];
            k0.q[1] = *(const uint4*)&Kc[m][ks * 32 + 16 * hi + 8];
            k1.q[0] = *(const uint4*)&Kc[16 + m][ks * 32 + 16 * hi];
            k1.q[1] = *(const uint4*)&Kc[16 + m][ks * 32 + 16 * hi + 8];
            s0 = wmma_bf16(qf[ks].v, k0.v, s0);
            s1 = wmma_bf16(qf[ks].v, k1.v, s1);
        }
        float b0 = biasCol[m], b1 = biasCol[16 + m];
#pragma unroll
        for (int r = 0; r < 8; ++r) {
            s0[r] = s0[r] * SCALE + b0;
            s1[r] = s1[r] * SCALE + b1;
        }
        // online softmax update per row (row = r + 8*hi, cols across 16 lanes)
#pragma unroll
        for (int r = 0; r < 8; ++r) {
            float mx = fmaxf(s0[r], s1[r]);
#pragma unroll
            for (int off = 8; off > 0; off >>= 1)
                mx = fmaxf(mx, __shfl_xor(mx, off, 32));
            float mnew = fmaxf(mstate[r], mx);
            float alpha = __expf(mstate[r] - mnew);
            mstate[r] = mnew;
            float p0 = __expf(s0[r] - mnew);
            float p1 = __expf(s1[r] - mnew);
            s0[r] = p0; s1[r] = p1;
            float rs = p0 + p1;
#pragma unroll
            for (int off = 8; off > 0; off >>= 1)
                rs += __shfl_xor(rs, off, 32);
            lstate[r] = lstate[r] * alpha + rs;
#pragma unroll
            for (int ot = 0; ot < 8; ++ot) oacc[ot][r] *= alpha;
        }
        // stage P (bf16) into per-wave LDS to re-enter WMMA A-layout
#pragma unroll
        for (int r = 0; r < 8; ++r) {
            int rr = r + 8 * hi;
            Pst[w][rr][m]      = (__bf16)s0[r];
            Pst[w][rr][16 + m] = (__bf16)s1[r];
        }
        __syncthreads();

        // O += P (16x32) * V (32x128)
        FragBF pf;
        pf.q[0] = *(const uint4*)&Pst[w][m][16 * hi];
        pf.q[1] = *(const uint4*)&Pst[w][m][16 * hi + 8];
#pragma unroll
        for (int ot = 0; ot < 8; ++ot) {
            FragBF vf;
            vf.q[0] = *(const uint4*)&Vt[ot * 16 + m][16 * hi];
            vf.q[1] = *(const uint4*)&Vt[ot * 16 + m][16 * hi + 8];
            oacc[ot] = wmma_bf16(pf.v, vf.v, oacc[ot]);
        }
    }

    // epilogue: normalize, scatter with forward q permutation -> (B,S,H,HD)
#pragma unroll
    for (int r = 0; r < 8; ++r) {
        float inv = 1.0f / lstate[r];
        size_t orow = ((size_t)b * SEQ + sqr[r]) * DMODEL + h * HDIM;
#pragma unroll
        for (int ot = 0; ot < 8; ++ot)
            Out[orow + ot * 16 + m] = (__bf16)(oacc[ot][r] * inv);
    }
}

// ---------------- workspace layout (bytes) ----------------
static constexpr size_t OFF_XB    = 0;                               // 64 MiB
static constexpr size_t OFF_WINB  = OFF_XB + (size_t)MROWS * DMODEL * 2;
static constexpr size_t OFF_WOUTB = OFF_WINB + (size_t)DMODEL * 3 * DMODEL * 2;
static constexpr size_t OFF_QKVB  = OFF_WOUTB + (size_t)DMODEL * DMODEL * 2;   // also FINAL alias
static constexpr size_t OFF_QB    = OFF_QKVB + (size_t)MROWS * 3 * DMODEL * 2;
static constexpr size_t OFF_KB    = OFF_QB + (size_t)BH * SEQ * HDIM * 2;
static constexpr size_t OFF_VB    = OFF_KB + (size_t)BH * SEQ * HDIM * 2;
static constexpr size_t OFF_HASH  = OFF_VB + (size_t)BH * SEQ * HDIM * 2;
static constexpr size_t OFF_SIDX  = OFF_HASH + (size_t)2 * BH * SEQ * 4;

extern "C" void kernel_launch(void* const* d_in, const int* in_sizes, int n_in,
                              void* d_out, int out_size, void* d_ws, size_t ws_size,
                              hipStream_t stream) {
    const float* x       = (const float*)d_in[0];
    const float* w_in    = (const float*)d_in[1];
    const float* w_out   = (const float*)d_in[2];
    const float* lshproj = (const float*)d_in[3];
    const int*   sampled = (const int*)d_in[4];
    float* out = (float*)d_out;

    char* ws = (char*)d_ws;
    __bf16* xb    = (__bf16*)(ws + OFF_XB);
    __bf16* winb  = (__bf16*)(ws + OFF_WINB);
    __bf16* woutb = (__bf16*)(ws + OFF_WOUTB);
    __bf16* qkvb  = (__bf16*)(ws + OFF_QKVB);
    __bf16* final_bf = (__bf16*)(ws + OFF_QKVB);   // reuse qkv region after RoPE
    __bf16* qb    = (__bf16*)(ws + OFF_QB);
    __bf16* kb    = (__bf16*)(ws + OFF_KB);
    __bf16* vb    = (__bf16*)(ws + OFF_VB);
    int* hashes   = (int*)(ws + OFF_HASH);
    int* sidx     = (int*)(ws + OFF_SIDX);
    int* sidx_q   = sidx;
    int* sidx_k   = sidx + (size_t)BH * SEQ;

    // 1) casts
    cast_f32_bf16_kernel<<<4096, 256, 0, stream>>>(x, xb, MROWS * DMODEL);
    cast_f32_bf16_kernel<<<4096, 256, 0, stream>>>(w_in, winb, DMODEL * 3 * DMODEL);
    cast_f32_bf16_kernel<<<2048, 256, 0, stream>>>(w_out, woutb, DMODEL * DMODEL);

    // 2) qkv = x @ w_in   (16384 x 6144, K=2048), bf16 out
    gemm_bf16_kernel<__bf16><<<dim3(3 * DMODEL / 128, MROWS / 128), 256, 0, stream>>>(
        xb, winb, qkvb, MROWS, 3 * DMODEL, DMODEL);

    // 3) RoPE + transpose to (B,H,S,HD)
    rope_kernel<<<(BATCH * SEQ * NHEAD * 64) / 256, 256, 0, stream>>>(qkvb, qb, kb, vb);

    // 4) LSH hashes for q (groups 0..31) and k (groups 32..63)
    lsh_hash_kernel<<<(2 * BH * SEQ) / 256, 256, 0, stream>>>(qb, kb, lshproj, hashes);

    // 5) stable argsort per (b,h) for q and k
    bitonic_sort_kernel<<<2 * BH, 1024, 0, stream>>>(hashes, sidx);

    // 6) fused block + sampled attention with LSE merge, scatter to (B,S,H,HD)
    hyper_attn_kernel<<<BH * NBLK, 512, 0, stream>>>(qb, kb, vb, sidx_q, sidx_k,
                                                     sampled, final_bf);

    // 7) out = final @ w_out  (16384 x 2048, K=2048), f32 out
    gemm_bf16_kernel<float><<<dim3(DMODEL / 128, MROWS / 128), 256, 0, stream>>>(
        final_bf, woutb, out, MROWS, DMODEL, DMODEL);
}
